// PairwiseFlowEncoder_43525198578383
// MI455X (gfx1250) — matrix operand order repaired and checked by reading, not compiled, not run on hardware
//
#include <hip/hip_runtime.h>
#include <hip/hip_bf16.h>

// ============================================================================
// PairwiseFlowEncoder on MI455X (gfx1250, wave32, WMMA).
//
// All 3x3 convs (BN folded, LeakyReLU fused) run as implicit GEMM on
// v_wmma_f32_16x16x32_f16: M = 16 output pixels, N = 16 output channels,
// K = 9 * Cin (padded to a multiple of 32, tap-major). Activations between
// layers are kept in f16. The A tile lives in LDS with ci innermost so each
// fragment half is 8 contiguous halves -> ds_load_b128 (no scalar gathers).
// Waves are N-blocked (NT accumulators) to reuse the A fragment across
// multiple WMMAs and amortize LDS staging.
// ============================================================================

typedef __attribute__((ext_vector_type(16))) _Float16 v16h;
typedef __attribute__((ext_vector_type(8)))  _Float16 v8h;
typedef __attribute__((ext_vector_type(8)))  float    v8f;

#define EPS_BN 1e-5f

// ---------------------------------------------------------------------------
// Fold BN into f16 weight matrix wmat[CoutP][9*CINP], K = tap*CINP + ci
// (tap-major so each tap contributes contiguous 32-wide K chunks).
// ---------------------------------------------------------------------------
__global__ void prep_weights_kernel(const float* __restrict__ w,
                                    const float* __restrict__ gamma,
                                    const float* __restrict__ beta,
                                    const float* __restrict__ mean,
                                    const float* __restrict__ var,
                                    const float* __restrict__ bias_in,
                                    int has_bn, int Cout, int Cin, int CINP, int CoutP,
                                    _Float16* __restrict__ wmat,
                                    float* __restrict__ biasv)
{
  const int KP = 9 * CINP;
  long long i = (long long)blockIdx.x * blockDim.x + threadIdx.x;
  long long total = (long long)CoutP * KP;
  if (i >= total) return;
  int k   = (int)(i % KP);
  int co  = (int)(i / KP);
  int tap = k / CINP;
  int ci  = k % CINP;
  float scale = 1.f, bv = 0.f;
  if (co < Cout) {
    if (has_bn) {
      scale = gamma[co] * rsqrtf(var[co] + EPS_BN);
      bv = beta[co] - mean[co] * scale;
    } else if (bias_in) {
      bv = bias_in[co];
    }
  }
  float v = 0.f;
  if (co < Cout && ci < Cin) v = w[(co * Cin + ci) * 9 + tap] * scale;
  wmat[i] = (_Float16)v;
  if (k == 0) biasv[co] = (co < Cout) ? bv : 0.f;
}

// ---------------------------------------------------------------------------
// Implicit-GEMM 3x3 conv, pad=1, stride 1 or 2, fused BN-bias + LeakyReLU.
// Block: 128 threads = 4 waves. Tile: 64 output pixels (contiguous along W,
// fixed b,ho) x (NT*16) output channels; wave w owns pixels [16w, 16w+16)
// and iterates NT channel tiles with NT accumulators (A-fragment reuse).
//
// LDS layouts (ci innermost => vectorized fragment loads):
//   aLds[(r*130 + x)*32 + ci]   : activation patch, f16, zero-padded
//   wLds[((co*9)+tap)*32 + ci]  : weight tile, co in [0, NT*16)
//
// A fragment (16x32 f16, contiguous K): lane L (hi = L>=16):
//   elems 0..7  = K 16*hi + 0..7     -> one 16B LDS load
//   elems 8..15 = K 16*hi + 8..15   -> one 16B LDS load
// B fragment (32x16 f16, same K order as A): lane L, N = L%16:
//   elems 0..15 = K 16*hi + 0..15    -> one 32B LDS load
// ---------------------------------------------------------------------------
template <typename TIN, int NT>
__global__ __launch_bounds__(128) void conv3x3_wmma_kernel(
    const TIN* __restrict__ in, int Cin, int CinTot, int cin_base, int Hin, int Win,
    const _Float16* __restrict__ wmat, const float* __restrict__ biasv,
    int Cout, int CINP, int stride, int lrelu,
    _Float16* __restrict__ out16, int c16b, int C16,
    float* __restrict__ out32, int c32b, int C32,
    int Hout, int Wout)
{
  __shared__ __align__(32) _Float16 aLds[3 * 130 * 32];   // 24,960 B
  __shared__ __align__(32) _Float16 wLds[64 * 9 * 32];    // 36,864 B (max NT=4)

  const int KP  = 9 * CINP;
  const int tid = threadIdx.x;

  const int wtiles = Wout >> 6;
  int t = blockIdx.x;
  const int wt = t % wtiles; t /= wtiles;
  const int ho = t % Hout;
  const int b  = t / Hout;
  const int wo0 = wt * 64;
  const int co0 = blockIdx.y * (16 * NT);

  const int wave = tid >> 5;
  const int lane = tid & 31;
  const int hi   = (lane & 16) ? 1 : 0;
  const int n    = lane & 15;               // output-channel column in D
  const int mx   = (wave * 16 + n) * stride; // pixel row (lane%16) * stride

  const int XW   = 64 * stride + 2;
  const int atot = 32 * 3 * XW;
  const int wtot = NT * 16 * 9 * 32;

  v8f acc[NT];
  #pragma unroll
  for (int nt = 0; nt < NT; ++nt) acc[nt] = (v8f){};

  for (int ci0 = 0; ci0 < CINP; ci0 += 32) {
    // ---- stage activation patch (x fastest per thread: coalesced global) ----
    for (int i = tid; i < atot; i += 128) {
      int x  = i % XW;
      int tt = i / XW;
      int r  = tt % 3;
      int ci = tt / 3;
      int gy = ho * stride - 1 + r;
      int gx = wo0 * stride - 1 + x;
      int cig = ci0 + ci;
      float v = 0.f;
      if (cig < Cin && (unsigned)gy < (unsigned)Hin && (unsigned)gx < (unsigned)Win)
        v = (float)in[(((size_t)b * CinTot + cin_base + cig) * Hin + gy) * Win + gx];
      aLds[((r * 130 + x) << 5) + ci] = (_Float16)v;   // ci-innermost (transposed)
    }
    // ---- stage weight tile (ci fastest: coalesced global, contiguous LDS) ----
    for (int i = tid; i < wtot; i += 128) {
      int ci  = i & 31;
      int tt  = i >> 5;
      int tap = tt % 9;
      int co  = tt / 9;
      wLds[i] = wmat[(size_t)(co0 + co) * KP + tap * CINP + ci0 + ci];
    }
    __syncthreads();

    #pragma unroll
    for (int r = 0; r < 3; ++r) {
      #pragma unroll
      for (int cc = 0; cc < 3; ++cc) {
        const int tap = r * 3 + cc;
        const int abase = ((r * 130 + mx + cc) << 5);
        v8h a0 = *(const v8h*)(aLds + abase + 16 * hi);        // K 16*hi + 0..7
        v8h a1 = *(const v8h*)(aLds + abase + 16 * hi + 8);    // K 16*hi + 8..15
        v16h va = __builtin_shufflevector(a0, a1,
            0, 1, 2, 3, 4, 5, 6, 7, 8, 9, 10, 11, 12, 13, 14, 15);
        #pragma unroll
        for (int nt = 0; nt < NT; ++nt) {
          const int co_l = nt * 16 + n;
          v16h vb = *(const v16h*)(wLds + (((co_l * 9) + tap) << 5) + 16 * hi);
          acc[nt] = __builtin_amdgcn_wmma_f32_16x16x32_f16(
              /*neg_a=*/false, va, /*neg_b=*/false, vb,
              /*c_mod=*/(short)0, acc[nt], /*reuse_a=*/false, /*reuse_b=*/false);
        }
      }
    }
    __syncthreads();
  }

  // ---- epilogue: D layout (lane -> N=lane%16; elem r -> M=r+8*(lane>=16)) ----
  #pragma unroll
  for (int nt = 0; nt < NT; ++nt) {
    const int co = co0 + nt * 16 + n;
    const float bs = biasv[co];
    #pragma unroll
    for (int r8 = 0; r8 < 8; ++r8) {
      int mm = wave * 16 + r8 + 8 * hi;
      int wo = wo0 + mm;
      float v = acc[nt][r8] + bs;
      if (lrelu) v = v > 0.f ? v : 0.1f * v;
      if (co < Cout) {
        if (out16) out16[(((size_t)b * C16 + c16b + co) * Hout + ho) * Wout + wo] = (_Float16)v;
        if (out32) out32[(((size_t)b * C32 + c32b + co) * Hout + ho) * Wout + wo] = v;
      }
    }
  }
}

// ---------------------------------------------------------------------------
// Inverse L2 norms over channels: invn[b,h,w] = 1/max(||f[b,:,h,w]||, 1e-12)
// ---------------------------------------------------------------------------
__global__ void invnorm_kernel(const float* __restrict__ feat, float* __restrict__ invn,
                               int C, int HW, long long total)
{
  long long i = (long long)blockIdx.x * blockDim.x + threadIdx.x;
  if (i >= total) return;
  int p = (int)(i % HW);
  int b = (int)(i / HW);
  float s = 0.f;
  for (int c = 0; c < C; ++c) {
    float f = feat[((size_t)b * C + c) * HW + p];
    s += f * f;
  }
  invn[i] = 1.f / fmaxf(sqrtf(s), 1e-12f);
}

// ---------------------------------------------------------------------------
// Dilated correlation: radius 4, dilations (1,2,4) -> 243 channels.
// f1/f2 read as f16 from the concat buffer (f1 @ ch0, f2 @ ch307 of 371).
// One block per (b, h); 128 threads = one per w. f1 row staged in LDS; f2
// served out of L2 (25 MB of f16 feature maps << 192 MB L2). Writes f32 to
// d_out and f16 into concat channels 64..306.
// ---------------------------------------------------------------------------
__global__ __launch_bounds__(128) void corr_kernel(
    _Float16* __restrict__ cat, const float* __restrict__ invn1,
    const float* __restrict__ invn2, float* __restrict__ corr_out)
{
  const int H = 96, W = 128, C = 64, CT = 371, F2 = 307, OC = 243;
  __shared__ _Float16 f1s[64 * 128];

  const int h = blockIdx.x % H;
  const int b = blockIdx.x / H;
  const int tid = threadIdx.x;

  for (int i = tid; i < C * W; i += 128) {
    int c = i >> 7, w = i & 127;
    f1s[i] = cat[(((size_t)b * CT + c) * H + h) * W + w];
  }
  __syncthreads();

  const int w = tid;
  const float inv1 = invn1[(b * H + h) * W + w];
  const _Float16* f2 = cat + ((size_t)b * CT + F2) * H * W;

  int och = 0;
  for (int di = 0; di < 3; ++di) {
    const int d = 1 << di;                         // dilations 1, 2, 4
    for (int dy = -4; dy <= 4; ++dy) {
      const int y2 = h + dy * d;
      for (int dx = -4; dx <= 4; ++dx, ++och) {
        const int x2 = w + dx * d;
        float dot = 0.f;
        if ((unsigned)y2 < (unsigned)H && (unsigned)x2 < (unsigned)W) {
          #pragma unroll 8
          for (int c = 0; c < C; ++c)
            dot += (float)f1s[(c << 7) + w] * (float)f2[((size_t)c * H + y2) * W + x2];
          dot *= inv1 * invn2[(b * H + y2) * W + x2];
        }
        corr_out[(((size_t)b * OC + och) * H + h) * W + w] = dot;
        cat[(((size_t)b * CT + 64 + och) * H + h) * W + w] = (_Float16)dot;
      }
    }
  }
}

// ===========================================================================
// Host-side orchestration
// ===========================================================================
namespace {

struct LayerDesc {
  const float *w, *gamma, *beta, *mean, *var, *bias;
  int has_bn, Cin, CINP, Cout, CoutP, stride, NT;
  _Float16* wmat;
  float* biasv;
};

inline size_t align256(size_t x) { return (x + 255) & ~(size_t)255; }

} // namespace

extern "C" void kernel_launch(void* const* d_in, const int* in_sizes, int n_in,
                              void* d_out, int out_size, void* d_ws, size_t ws_size,
                              hipStream_t stream) {
  (void)in_sizes; (void)n_in; (void)out_size; (void)ws_size;

  // --- input mapping (jax pytree flatten order: dict keys sorted) ---------
  // 0: img1, 1: img2
  // 2+5i .. 6+5i : backbone[i] {beta, gamma, mean, var, w}, i = 0..5
  // 32: flow_b
  // 33..37: flow_blk {beta, gamma, mean, var, w}
  // 38: flow_w
  // 39..43: fuse[0], 44..48: fuse[1]
  const float* img1 = (const float*)d_in[0];
  const float* img2 = (const float*)d_in[1];

  auto bn5 = [&](int base, LayerDesc& L) {
    L.beta  = (const float*)d_in[base + 0];
    L.gamma = (const float*)d_in[base + 1];
    L.mean  = (const float*)d_in[base + 2];
    L.var   = (const float*)d_in[base + 3];
    L.w     = (const float*)d_in[base + 4];
    L.bias  = nullptr;
    L.has_bn = 1;
  };

  LayerDesc Ls[10] = {};
  // backbone
  static const int bbCin[6]  = { 3, 16, 16, 32, 32, 64};
  static const int bbCinP[6] = {32, 32, 32, 32, 32, 64};
  static const int bbCout[6] = {16, 16, 32, 32, 64, 64};
  static const int bbStr[6]  = { 2,  1,  2,  1,  2,  1};
  static const int bbNT[6]   = { 1,  1,  2,  2,  4,  4};
  for (int i = 0; i < 6; ++i) {
    bn5(2 + 5 * i, Ls[i]);
    Ls[i].Cin = bbCin[i]; Ls[i].CINP = bbCinP[i];
    Ls[i].Cout = bbCout[i]; Ls[i].CoutP = bbCout[i];
    Ls[i].stride = bbStr[i]; Ls[i].NT = bbNT[i];
  }
  // fuse[0]: 371 -> 128
  bn5(39, Ls[6]); Ls[6].Cin = 371; Ls[6].CINP = 384; Ls[6].Cout = 128; Ls[6].CoutP = 128; Ls[6].stride = 1; Ls[6].NT = 4;
  // fuse[1]: 128 -> 128
  bn5(44, Ls[7]); Ls[7].Cin = 128; Ls[7].CINP = 128; Ls[7].Cout = 128; Ls[7].CoutP = 128; Ls[7].stride = 1; Ls[7].NT = 4;
  // flow_blk: 128 -> 64
  bn5(33, Ls[8]); Ls[8].Cin = 128; Ls[8].CINP = 128; Ls[8].Cout = 64; Ls[8].CoutP = 64; Ls[8].stride = 1; Ls[8].NT = 4;
  // flow conv: 64 -> 2 (bias, no BN, no relu)
  Ls[9].w = (const float*)d_in[38]; Ls[9].bias = (const float*)d_in[32];
  Ls[9].gamma = Ls[9].beta = Ls[9].mean = Ls[9].var = nullptr;
  Ls[9].has_bn = 0; Ls[9].Cin = 64; Ls[9].CINP = 64; Ls[9].Cout = 2; Ls[9].CoutP = 16; Ls[9].stride = 1; Ls[9].NT = 1;

  // --- workspace carve ----------------------------------------------------
  size_t off = 0;
  auto carve = [&](size_t bytes) -> void* {
    off = align256(off);
    void* p = (char*)d_ws + off;
    off += bytes;
    return p;
  };
  for (int l = 0; l < 10; ++l) {
    size_t kp = (size_t)9 * Ls[l].CINP;
    Ls[l].wmat  = (_Float16*)carve((size_t)Ls[l].CoutP * kp * sizeof(_Float16));
    Ls[l].biasv = (float*)carve((size_t)Ls[l].CoutP * sizeof(float));
  }
  const size_t ACT_ELEMS = (size_t)8 * 16 * 384 * 512;          // largest layer out
  _Float16* actA   = (_Float16*)carve(ACT_ELEMS * sizeof(_Float16));
  _Float16* actB   = (_Float16*)carve(ACT_ELEMS * sizeof(_Float16));
  _Float16* catbuf = (_Float16*)carve((size_t)8 * 371 * 96 * 128 * sizeof(_Float16));
  float* invn1 = (float*)carve((size_t)8 * 96 * 128 * sizeof(float));
  float* invn2 = (float*)carve((size_t)8 * 96 * 128 * sizeof(float));

  // --- output regions (tuple concatenated flat) ---------------------------
  float* out = (float*)d_out;
  float* feat_out[2] = { out + 0, out + 6291456 };      // 8*64*96*128 each
  float* corr_out = out + 12582912;                     // 8*243*96*128
  float* pair_out = out + 36470784;                     // 8*128*96*128
  float* flow_out = out + 49053696;                     // 8*2*96*128

  // --- 1) fold BN into f16 weights ---------------------------------------
  for (int l = 0; l < 10; ++l) {
    long long tot = (long long)Ls[l].CoutP * 9 * Ls[l].CINP;
    int blocks = (int)((tot + 255) / 256);
    prep_weights_kernel<<<blocks, 256, 0, stream>>>(
        Ls[l].w, Ls[l].gamma, Ls[l].beta, Ls[l].mean, Ls[l].var, Ls[l].bias,
        Ls[l].has_bn, Ls[l].Cout, Ls[l].Cin, Ls[l].CINP, Ls[l].CoutP,
        Ls[l].wmat, Ls[l].biasv);
  }

  // --- conv launcher ------------------------------------------------------
  auto conv = [&](const void* in, bool f32in, int Cin, int CinTot, int cinb,
                  int Hin, int Win, int li,
                  _Float16* o16, int c16b, int C16,
                  float* o32, int c32b, int C32, bool lr) {
    const LayerDesc& L = Ls[li];
    int Hout = (Hin - 1) / L.stride + 1;
    int Wout = (Win - 1) / L.stride + 1;
    dim3 grid((unsigned)((Wout / 64) * Hout * 8), (unsigned)(L.CoutP / (16 * L.NT)));
    int lri = lr ? 1 : 0;
    if (f32in) {
      // only NT=1 used with f32 input (first backbone layer)
      conv3x3_wmma_kernel<float, 1><<<grid, 128, 0, stream>>>(
          (const float*)in, Cin, CinTot, cinb, Hin, Win, L.wmat, L.biasv,
          L.Cout, L.CINP, L.stride, lri, o16, c16b, C16, o32, c32b, C32, Hout, Wout);
    } else if (L.NT == 1) {
      conv3x3_wmma_kernel<_Float16, 1><<<grid, 128, 0, stream>>>(
          (const _Float16*)in, Cin, CinTot, cinb, Hin, Win, L.wmat, L.biasv,
          L.Cout, L.CINP, L.stride, lri, o16, c16b, C16, o32, c32b, C32, Hout, Wout);
    } else if (L.NT == 2) {
      conv3x3_wmma_kernel<_Float16, 2><<<grid, 128, 0, stream>>>(
          (const _Float16*)in, Cin, CinTot, cinb, Hin, Win, L.wmat, L.biasv,
          L.Cout, L.CINP, L.stride, lri, o16, c16b, C16, o32, c32b, C32, Hout, Wout);
    } else {
      conv3x3_wmma_kernel<_Float16, 4><<<grid, 128, 0, stream>>>(
          (const _Float16*)in, Cin, CinTot, cinb, Hin, Win, L.wmat, L.biasv,
          L.Cout, L.CINP, L.stride, lri, o16, c16b, C16, o32, c32b, C32, Hout, Wout);
    }
  };

  // --- 2) backbone on both images; feat written straight into concat ------
  for (int im = 0; im < 2; ++im) {
    const float* src = im ? img2 : img1;
    conv(src,  true, 3,  3,  0, 768, 1024, 0, actA, 0, 16, nullptr, 0, 0, true);
    conv(actA, false, 16, 16, 0, 384, 512, 1, actB, 0, 16, nullptr, 0, 0, true);
    conv(actB, false, 16, 16, 0, 384, 512, 2, actA, 0, 32, nullptr, 0, 0, true);
    conv(actA, false, 32, 32, 0, 192, 256, 3, actB, 0, 32, nullptr, 0, 0, true);
    conv(actB, false, 32, 32, 0, 192, 256, 4, actA, 0, 64, nullptr, 0, 0, true);
    conv(actA, false, 64, 64, 0, 96, 128, 5,
         catbuf, im ? 307 : 0, 371,            // f16 -> concat channels
         feat_out[im], 0, 64, true);           // f32 -> d_out feat region
  }

  // --- 3) correlation -----------------------------------------------------
  invnorm_kernel<<<384, 256, 0, stream>>>(feat_out[0], invn1, 64, 96 * 128, 8LL * 96 * 128);
  invnorm_kernel<<<384, 256, 0, stream>>>(feat_out[1], invn2, 64, 96 * 128, 8LL * 96 * 128);
  corr_kernel<<<8 * 96, 128, 0, stream>>>(catbuf, invn1, invn2, corr_out);

  // --- 4) fuse + flow head ------------------------------------------------
  conv(catbuf, false, 371, 371, 0, 96, 128, 6, actA, 0, 128, nullptr, 0, 0, true);
  conv(actA,   false, 128, 128, 0, 96, 128, 7, actB, 0, 128, pair_out, 0, 128, true);
  conv(actB,   false, 128, 128, 0, 96, 128, 8, actA, 0, 64, nullptr, 0, 0, true);
  conv(actA,   false, 64,  64,  0, 96, 128, 9, nullptr, 0, 0, flow_out, 0, 2, false);
}
